// LaneGcnNeck_52776558133724
// MI455X (gfx1250) — matrix operand order repaired and checked by reading, not compiled, not run on hardware
//
#include <hip/hip_runtime.h>
#include <stdint.h>

// ---------------------------------------------------------------------------
// LaneGCN neck for MI455X (gfx1250, wave32). All matmuls run through
// v_wmma_f32_16x16x32_bf16. f32 inputs and weights are converted to bf16
// while staging into LDS (conversion out of the hot loop); the k-loop is
// compile-time sized (KPAD template) and fully unrolled: inner loop is
// LDS fragment reads (b128 for A, 16-bit elements for B) + v_wmma only. LDS tiles use padded pitches
// to avoid same-bank strides. GroupNorm / ReLU / residual / scatter-add are
// fused into the GEMM epilogue. Edge lists (filter_distance) are built
// deterministically in row-major order via count -> scan -> ballot write.
// ---------------------------------------------------------------------------

#define N_MAP   20000
#define N_ACTS  1200
#define NF      128
#define E_A2M   16384
#define E_M2A   12288
#define E_A2A   98304
#define E_LR    4096
#define GN_EPS  1e-5f
#define SWP     136        // sW pitch (ushorts): 272B rows -> 4-bank shift/k
#define SOP     132        // sO pitch (floats)

// ---------------- bf16 helpers -------------------------------------------
typedef __bf16 bf16_t;
typedef bf16_t v16bf __attribute__((ext_vector_type(16)));
typedef float  v8f   __attribute__((ext_vector_type(8)));
typedef unsigned short u16x8 __attribute__((ext_vector_type(8)));

union V16 { v16bf bf; u16x8 u[2]; };

__device__ __forceinline__ unsigned short f2bf_bits(float f) {
  unsigned u = __builtin_bit_cast(unsigned, f);
  u += 0x7FFFu + ((u >> 16) & 1u);          // round-to-nearest-even
  return (unsigned short)(u >> 16);
}

// ---------------------------------------------------------------------------
// Generic fused GEMM: OUT[M,128] = X[M,Ktot] @ W[128,Ktot]^T  (+ epilogue)
// X assembled from up to 3 segments, each optionally row-gathered via index.
// GN: per-row GroupNorm(1) affine. ACT: ReLU. RES: += res then ReLU.
// SCATTER: atomicAdd rows into out[oidx[r]]. If eCnt != null, effective
// M = min(M, *eCnt) (edge count computed on device).
// ---------------------------------------------------------------------------
template<int KPAD, bool GN, bool ACT, bool RES, bool SCATTER>
__global__ void __launch_bounds__(128)
gemm128_kernel(const float* __restrict__ X0, const int* __restrict__ I0, int k0,
               const float* __restrict__ X1, const int* __restrict__ I1, int k1,
               const float* __restrict__ X2, const int* __restrict__ I2, int k2,
               const float* __restrict__ W,
               const float* __restrict__ gw, const float* __restrict__ gb,
               const float* __restrict__ res,
               const int*   __restrict__ oidx,
               float* __restrict__ out,
               int M, const int* __restrict__ eCnt)
{
  constexpr int XP = KPAD + 8;              // sX pitch (ushorts)
  __shared__ unsigned short sX[16 * XP];    // X tile, bf16 bits
  __shared__ unsigned short sW[KPAD * SWP]; // W^T tile, bf16 bits: sW[k][o]
  __shared__ float sO[16 * SOP];            // f32 output tile
  __shared__ float sMean[16];
  __shared__ float sRstd[16];

  const int tid = threadIdx.x;
  int Meff = M;
  if (eCnt) { int e = *eCnt; if (e < Meff) Meff = e; }
  const int tile = blockIdx.x * 16;
  if (tile >= Meff) return;                 // block-uniform exit (EXEC stays full)
  int rowsA = Meff - tile; if (rowsA > 16) rowsA = 16;

  const int Ktot = k0 + k1 + k2;            // == KPAD or KPAD-rounded (Ktot%4==0)

  // ---- stage W transposed as bf16: sW[k*SWP + o], o == tid ---------------
  {
    const float* wrow = W + (long)tid * Ktot;
    for (int k4 = 0; k4 < Ktot; k4 += 4) {
      float4 wv = *reinterpret_cast<const float4*>(wrow + k4);
      sW[(k4 + 0) * SWP + tid] = f2bf_bits(wv.x);
      sW[(k4 + 1) * SWP + tid] = f2bf_bits(wv.y);
      sW[(k4 + 2) * SWP + tid] = f2bf_bits(wv.z);
      sW[(k4 + 3) * SWP + tid] = f2bf_bits(wv.w);
    }
    for (int k = Ktot; k < KPAD; ++k) sW[k * SWP + tid] = 0;
  }

  // ---- stage X tile (f32 -> bf16), zero-fill padding and inactive rows ---
  for (int r = 0; r < 16; ++r) {
    const int  gr  = tile + r;
    const bool act = (r < rowsA);
    for (int k = tid; k < KPAD; k += 128) {
      float v = 0.f;
      if (act && k < Ktot) {
        if (k < k0)            { int rr = I0 ? I0[gr] : gr; v = X0[(long)rr * k0 + k]; }
        else if (k < k0 + k1)  { int rr = I1 ? I1[gr] : gr; v = X1[(long)rr * k1 + (k - k0)]; }
        else                   { int rr = I2 ? I2[gr] : gr; v = X2[(long)rr * k2 + (k - k0 - k1)]; }
      }
      sX[r * XP + k] = f2bf_bits(v);
    }
  }
  __syncthreads();

  // ---- WMMA: 4 waves x 2 n-tiles = 128 output columns --------------------
  const int lane  = tid & 31;
  const int nlo   = lane & 15;              // A: M row / B,D: N col (low)
  const int khalf = lane >> 4;
  const int o0    = (tid >> 5) * 32;        // this wave's first output column

  v8f acc0 = {};
  v8f acc1 = {};

#pragma unroll
  for (int kb = 0; kb < KPAD; kb += 32) {
    // A fragment (16x32): lane m=nlo; elems 0..7 -> k=kb+khalf*8.., 8..15 -> +16
    V16 a;
    a.u[0] = *reinterpret_cast<const u16x8*>(sX + nlo * XP + kb + khalf * 8);
    a.u[1] = *reinterpret_cast<const u16x8*>(sX + nlo * XP + kb + 16 + khalf * 8);
    // B fragments (32x16): lane col N = nlo; elems 0..7 -> k=kb+khalf*8.., 8..15 -> +16
    V16 b0, b1;
#pragma unroll
    for (int i = 0; i < 8; ++i) {
      const int r0 = (kb + khalf * 8 + i) * SWP + o0 + nlo;
      const int r1 = (kb + 16 + khalf * 8 + i) * SWP + o0 + nlo;
      b0.u[0][i] = sW[r0];
      b0.u[1][i] = sW[r1];
      b1.u[0][i] = sW[r0 + 16];
      b1.u[1][i] = sW[r1 + 16];
    }
    acc0 = __builtin_amdgcn_wmma_f32_16x16x32_bf16(false, a.bf, false, b0.bf, (short)0, acc0, false, false);
    acc1 = __builtin_amdgcn_wmma_f32_16x16x32_bf16(false, a.bf, false, b1.bf, (short)0, acc1, false, false);
  }

  // ---- spill accumulators to LDS (D layout: vgpr r -> M = r + 8*khalf) ---
#pragma unroll
  for (int r = 0; r < 8; ++r) {
    int row = r + khalf * 8;
    sO[row * SOP + o0 + nlo]      = acc0[r];
    sO[row * SOP + o0 + 16 + nlo] = acc1[r];
  }
  __syncthreads();

  if (GN) {
    if (tid < 16) {
      float s = 0.f, s2 = 0.f;
      for (int c = 0; c < NF; ++c) { float v = sO[tid * SOP + c]; s += v; s2 += v * v; }
      float mean = s * (1.f / NF);
      float var  = s2 * (1.f / NF) - mean * mean;
      sMean[tid] = mean;
      sRstd[tid] = rsqrtf(var + GN_EPS);
    }
    __syncthreads();
  }

  const int col = tid;                       // 0..127, coalesced stores
  for (int r = 0; r < rowsA; ++r) {
    float v = sO[r * SOP + col];
    if (GN)  v = (v - sMean[r]) * sRstd[r] * gw[col] + gb[col];
    if (RES) v += res[(long)(tile + r) * NF + col];
    if (ACT) v = fmaxf(v, 0.f);
    if (SCATTER) atomicAdd(out + (long)oidx[tile + r] * NF + col, v);
    else         out[(long)(tile + r) * NF + col] = v;
  }
}

// ---- dist0: d = relu(dist @ W0^T + b0), K=2 (too thin for WMMA) ----------
__global__ void __launch_bounds__(128)
dist0_kernel(const float* __restrict__ dist, const float* __restrict__ w0,
             const float* __restrict__ b0, float* __restrict__ out,
             const int* __restrict__ eCnt)
{
  const int e = blockIdx.x;
  if (e >= *eCnt) return;
  const int o = threadIdx.x;
  float dx = dist[e * 2 + 0], dy = dist[e * 2 + 1];
  out[(long)e * NF + o] = fmaxf(dx * w0[o * 2 + 0] + dy * w0[o * 2 + 1] + b0[o], 0.f);
}

// ---- standalone GroupNorm+ReLU (post scatter-add) ------------------------
__global__ void __launch_bounds__(128)
gn_relu_kernel(const float* __restrict__ x, const float* __restrict__ gw,
               const float* __restrict__ gb, float* __restrict__ out)
{
  const int r = blockIdx.x;
  const int t = threadIdx.x;
  __shared__ float s1[128], s2[128];
  float v = x[(long)r * NF + t];
  s1[t] = v; s2[t] = v * v;
  __syncthreads();
  for (int d = 64; d > 0; d >>= 1) {
    if (t < d) { s1[t] += s1[t + d]; s2[t] += s2[t + d]; }
    __syncthreads();
  }
  float mean = s1[0] * (1.f / NF);
  float var  = s2[0] * (1.f / NF) - mean * mean;
  float rstd = rsqrtf(var + GN_EPS);
  out[(long)r * NF + t] = fmaxf((v - mean) * rstd * gw[t] + gb[t], 0.f);
}

// ---- concat(nodes, turn, control, intersect) -> [N_MAP, 132] -------------
__global__ void concat_meta_kernel(const float* __restrict__ nodes,
                                   const float* __restrict__ turn,
                                   const float* __restrict__ ctrl,
                                   const float* __restrict__ inter,
                                   float* __restrict__ nm)
{
  int i = blockIdx.x * blockDim.x + threadIdx.x;
  if (i >= N_MAP * 132) return;
  int n = i / 132, k = i - n * 132;
  float v;
  if (k < 128)      v = nodes[n * NF + k];
  else if (k < 130) v = turn[n * 2 + (k - 128)];
  else if (k == 130) v = ctrl[n];
  else              v = inter[n];
  nm[i] = v;
}

// ---- filter_distance: per-row counts (one wave32 per agt row) ------------
__global__ void __launch_bounds__(128)
edge_count_kernel(const float* __restrict__ actr, const float* __restrict__ cctr,
                  int Na, int Nc, float th, int* __restrict__ counts)
{
  int row = blockIdx.x * 4 + (threadIdx.x >> 5);
  if (row >= Na) return;
  int lane = threadIdx.x & 31;
  float ax = actr[row * 2], ay = actr[row * 2 + 1];
  int cnt = 0;
  for (int base = 0; base < Nc; base += 32) {
    int c = base + lane;
    bool hit = false;
    if (c < Nc) {
      float dx = ax - cctr[c * 2], dy = ay - cctr[c * 2 + 1];
      hit = sqrtf(dx * dx + dy * dy) <= th;
    }
    unsigned m = (unsigned)__ballot(hit);
    cnt += __popc(m);
  }
  if (lane == 0) counts[row] = cnt;
}

// ---- single-block exclusive scan + clamped total -------------------------
__global__ void __launch_bounds__(1024)
scan_kernel(const int* __restrict__ counts, int* __restrict__ offs,
            int* __restrict__ eCnt, int n, int maxE)
{
  __shared__ int s[1024];
  __shared__ int carry;
  int t = threadIdx.x;
  if (t == 0) carry = 0;
  __syncthreads();
  for (int base = 0; base < n; base += 1024) {
    int v = (base + t < n) ? counts[base + t] : 0;
    s[t] = v;
    __syncthreads();
    for (int d = 1; d < 1024; d <<= 1) {
      int add = (t >= d) ? s[t - d] : 0;
      __syncthreads();
      s[t] += add;
      __syncthreads();
    }
    int base_off = carry;
    if (base + t < n) offs[base + t] = base_off + s[t] - v;
    int tot = s[1023];
    __syncthreads();
    if (t == 0) carry = base_off + tot;
    __syncthreads();
  }
  if (t == 0) eCnt[0] = (carry < maxE) ? carry : maxE;
}

// ---- edge write: row-major order matching jnp.nonzero --------------------
__global__ void __launch_bounds__(128)
edge_write_kernel(const float* __restrict__ actr, const float* __restrict__ cctr,
                  int Na, int Nc, float th, const int* __restrict__ offs, int maxE,
                  int* __restrict__ hi, int* __restrict__ wi, float* __restrict__ dist)
{
  int row = blockIdx.x * 4 + (threadIdx.x >> 5);
  if (row >= Na) return;
  int lane = threadIdx.x & 31;
  float ax = actr[row * 2], ay = actr[row * 2 + 1];
  int pos = offs[row];
  unsigned lt = (1u << lane) - 1u;
  for (int base = 0; base < Nc; base += 32) {
    int c = base + lane;
    bool hit = false; float dx = 0.f, dy = 0.f;
    if (c < Nc) {
      dx = ax - cctr[c * 2]; dy = ay - cctr[c * 2 + 1];
      hit = sqrtf(dx * dx + dy * dy) <= th;
    }
    unsigned m = (unsigned)__ballot(hit);
    if (hit) {
      int p = pos + __popc(m & lt);
      if (p < maxE) {
        hi[p] = row; wi[p] = c;
        dist[p * 2 + 0] = dx; dist[p * 2 + 1] = dy;
      }
    }
    pos += __popc(m);
  }
}

// ===========================================================================
// Host orchestration
// ===========================================================================
namespace {

struct Ws {
  float *nm, *nb0, *nb1, *nb2, *ab0, *ab1, *ab2;
  float *EB0, *EB1, *EB2, *EB3, *dist;
  int *hi, *wi, *counts, *offs, *eCnt;
};

static Ws carve(void* base) {
  char* c = (char*)base;
  auto take = [&](size_t bytes) -> void* {
    void* p = (void*)c;
    c += (bytes + 255) & ~(size_t)255;
    return p;
  };
  Ws w;
  w.nm  = (float*)take((size_t)N_MAP * 132 * 4);
  w.nb0 = (float*)take((size_t)N_MAP * NF * 4);
  w.nb1 = (float*)take((size_t)N_MAP * NF * 4);
  w.nb2 = (float*)take((size_t)N_MAP * NF * 4);
  w.ab0 = (float*)take((size_t)N_ACTS * NF * 4);
  w.ab1 = (float*)take((size_t)N_ACTS * NF * 4);
  w.ab2 = (float*)take((size_t)N_ACTS * NF * 4);
  w.EB0 = (float*)take((size_t)E_A2A * NF * 4);
  w.EB1 = (float*)take((size_t)E_A2A * NF * 4);
  w.EB2 = (float*)take((size_t)E_A2A * NF * 4);
  w.EB3 = (float*)take((size_t)E_A2A * NF * 4);
  w.dist   = (float*)take((size_t)E_A2A * 2 * 4);
  w.hi     = (int*)take((size_t)E_A2A * 4);
  w.wi     = (int*)take((size_t)E_A2A * 4);
  w.counts = (int*)take((size_t)N_MAP * 4);
  w.offs   = (int*)take((size_t)N_MAP * 4);
  w.eCnt   = (int*)take(256);
  return w;
}

// Att-layer params. Leaf order assumes jax tree_flatten (sorted dict keys):
// agt_w, ctx0.{gb,gw,w}, ctx1_w, dist0_b, dist0_w, dist1.{gb,gw,w},
// linear.{gb,gw,w}, norm_gb, norm_gw, query.{gb,gw,w}  -> 18 leaves.
struct AttP {
  const float *agt_w, *ctx0_gb, *ctx0_gw, *ctx0_w, *ctx1_w;
  const float *dist0_b, *dist0_w;
  const float *dist1_gb, *dist1_gw, *dist1_w;
  const float *lin_gb, *lin_gw, *lin_w;
  const float *norm_gb, *norm_gw;
  const float *q_gb, *q_gw, *q_w;
};
static AttP load_attp(void* const* d_in, int b) {
  AttP p;
  p.agt_w    = (const float*)d_in[b + 0];
  p.ctx0_gb  = (const float*)d_in[b + 1];
  p.ctx0_gw  = (const float*)d_in[b + 2];
  p.ctx0_w   = (const float*)d_in[b + 3];
  p.ctx1_w   = (const float*)d_in[b + 4];
  p.dist0_b  = (const float*)d_in[b + 5];
  p.dist0_w  = (const float*)d_in[b + 6];
  p.dist1_gb = (const float*)d_in[b + 7];
  p.dist1_gw = (const float*)d_in[b + 8];
  p.dist1_w  = (const float*)d_in[b + 9];
  p.lin_gb   = (const float*)d_in[b + 10];
  p.lin_gw   = (const float*)d_in[b + 11];
  p.lin_w    = (const float*)d_in[b + 12];
  p.norm_gb  = (const float*)d_in[b + 13];
  p.norm_gw  = (const float*)d_in[b + 14];
  p.q_gb     = (const float*)d_in[b + 15];
  p.q_gw     = (const float*)d_in[b + 16];
  p.q_w      = (const float*)d_in[b + 17];
  return p;
}

enum GMode { G_PLAIN, G_GN_ACT, G_GN_RES_ACT, G_SCATTER };

struct GArgs {
  const float* X0 = nullptr; const int* I0 = nullptr; int k0 = 0;
  const float* X1 = nullptr; const int* I1 = nullptr; int k1 = 0;
  const float* X2 = nullptr; const int* I2 = nullptr; int k2 = 0;
  const float* W  = nullptr;
  const float* gw = nullptr; const float* gb = nullptr;
  const float* res = nullptr;
  const int*   oidx = nullptr;
  float* out = nullptr;
  int M = 0;
  const int* eCnt = nullptr;
};

#define GEMM_CALL(KP, GN, ACT, RES, SC)                                        \
  gemm128_kernel<KP, GN, ACT, RES, SC><<<grid, 128, 0, s>>>(                   \
      a.X0, a.I0, a.k0, a.X1, a.I1, a.k1, a.X2, a.I2, a.k2,                    \
      a.W, a.gw, a.gb, a.res, a.oidx, a.out, a.M, a.eCnt)

template<int KP>
static void launch_gemm_k(hipStream_t s, GMode mode, const GArgs& a) {
  int grid = (a.M + 15) / 16;
  switch (mode) {
    case G_PLAIN:      GEMM_CALL(KP, false, false, false, false); break;
    case G_GN_ACT:     GEMM_CALL(KP, true,  true,  false, false); break;
    case G_GN_RES_ACT: GEMM_CALL(KP, true,  true,  true,  false); break;
    case G_SCATTER:    GEMM_CALL(KP, false, false, false, true ); break;
  }
}

static void launch_gemm(hipStream_t s, GMode mode, const GArgs& a) {
  const int ktot = a.k0 + a.k1 + a.k2;
  const int kp   = (ktot + 31) & ~31;
  if (kp == 128)      launch_gemm_k<128>(s, mode, a);
  else if (kp == 160) launch_gemm_k<160>(s, mode, a);
  else                launch_gemm_k<384>(s, mode, a);
}

static void build_edges(hipStream_t s, Ws& w, const float* agt, int Na,
                        const float* ctx, int Nc, float th, int Emax) {
  edge_count_kernel<<<(Na + 3) / 4, 128, 0, s>>>(agt, ctx, Na, Nc, th, w.counts);
  scan_kernel<<<1, 1024, 0, s>>>(w.counts, w.offs, w.eCnt, Na, Emax);
  edge_write_kernel<<<(Na + 3) / 4, 128, 0, s>>>(agt, ctx, Na, Nc, th, w.offs,
                                                 Emax, w.hi, w.wi, w.dist);
}

// One LaneGCN Att layer. agts:[Na,128], ctx:[*,128]; edges in w.{hi,wi,dist}.
static void run_att(hipStream_t s, Ws& w, const AttP& p,
                    const float* agts, int Na, const float* ctx,
                    int Emax, float* accum, float* gbuf, float* outb) {
  // d = relu(dist @ dist0^T + b)
  dist0_kernel<<<Emax, 128, 0, s>>>(w.dist, p.dist0_w, p.dist0_b, w.EB0, w.eCnt);
  // d = lin_gn(d, dist1)
  { GArgs a; a.X0 = w.EB0; a.k0 = NF; a.W = p.dist1_w; a.gw = p.dist1_gw;
    a.gb = p.dist1_gb; a.out = w.EB1; a.M = Emax; a.eCnt = w.eCnt;
    launch_gemm(s, G_GN_ACT, a); }
  // q = lin_gn(agts[hi], query)
  { GArgs a; a.X0 = agts; a.I0 = w.hi; a.k0 = NF; a.W = p.q_w; a.gw = p.q_gw;
    a.gb = p.q_gb; a.out = w.EB2; a.M = Emax; a.eCnt = w.eCnt;
    launch_gemm(s, G_GN_ACT, a); }
  // c = lin_gn(concat(d, q, ctx[wi]), ctx0)   (virtual concat, K=384)
  { GArgs a; a.X0 = w.EB1; a.k0 = NF; a.X1 = w.EB2; a.k1 = NF;
    a.X2 = ctx; a.I2 = w.wi; a.k2 = NF;
    a.W = p.ctx0_w; a.gw = p.ctx0_gw; a.gb = p.ctx0_gb;
    a.out = w.EB3; a.M = Emax; a.eCnt = w.eCnt;
    launch_gemm(s, G_GN_ACT, a); }
  // accum = agts @ agt_w^T  (dense)
  { GArgs a; a.X0 = agts; a.k0 = NF; a.W = p.agt_w; a.out = accum; a.M = Na;
    launch_gemm(s, G_PLAIN, a); }
  // accum[hi] += c @ ctx1^T  (fused GEMM + atomic scatter; invalid edges skipped)
  { GArgs a; a.X0 = w.EB3; a.k0 = NF; a.W = p.ctx1_w; a.oidx = w.hi;
    a.out = accum; a.M = Emax; a.eCnt = w.eCnt;
    launch_gemm(s, G_SCATTER, a); }
  // gbuf = relu(gn(accum))
  gn_relu_kernel<<<Na, 128, 0, s>>>(accum, p.norm_gw, p.norm_gb, gbuf);
  // outb = relu(GN(gbuf @ lin_w^T) + agts)
  { GArgs a; a.X0 = gbuf; a.k0 = NF; a.W = p.lin_w; a.gw = p.lin_gw;
    a.gb = p.lin_gb; a.res = agts; a.out = outb; a.M = Na;
    launch_gemm(s, G_GN_RES_ACT, a); }
}

} // namespace

extern "C" void kernel_launch(void* const* d_in, const int* in_sizes, int n_in,
                              void* d_out, int out_size, void* d_ws, size_t ws_size,
                              hipStream_t s) {
  (void)in_sizes; (void)n_in; (void)out_size; (void)ws_size;

  const float* actors      = (const float*)d_in[0];
  const float* nodes       = (const float*)d_in[1];
  const float* actor_ctrs  = (const float*)d_in[2];
  const float* graph_ctrs  = (const float*)d_in[3];
  const float* graph_turn  = (const float*)d_in[4];
  const float* graph_ctrl  = (const float*)d_in[5];
  const float* graph_inter = (const float*)d_in[6];
  const int*   pre_u  = (const int*)d_in[7];
  const int*   pre_v  = (const int*)d_in[8];
  const int*   suc_u  = (const int*)d_in[9];
  const int*   suc_v  = (const int*)d_in[10];
  const int*   left_u = (const int*)d_in[11];
  const int*   left_v = (const int*)d_in[12];
  const int*   right_u = (const int*)d_in[13];
  const int*   right_v = (const int*)d_in[14];

  // params flattened jax-style at d_in[15..]: a2a.att[0,1] | a2m.att[0,1],
  // a2m.meta.{gb,gw,w} | m2a.att[0,1] | m2m.{ctr,ctr2,left,norm_gb,norm_gw,pre,right,suc}
  const int PB = 15;
  AttP a2a0 = load_attp(d_in, PB + 0);
  AttP a2a1 = load_attp(d_in, PB + 18);
  AttP a2m0 = load_attp(d_in, PB + 36);
  AttP a2m1 = load_attp(d_in, PB + 54);
  const float* meta_gb = (const float*)d_in[PB + 72];
  const float* meta_gw = (const float*)d_in[PB + 73];
  const float* meta_w  = (const float*)d_in[PB + 74];
  AttP m2a0 = load_attp(d_in, PB + 75);
  AttP m2a1 = load_attp(d_in, PB + 93);
  const int MB = PB + 111;   // m2m base (== 126)
  auto m2m_ctr     = [&](int i) { return (const float*)d_in[MB + i]; };
  auto m2m_ctr2_gb = [&](int i) { return (const float*)d_in[MB + 4 + 3 * i + 0]; };
  auto m2m_ctr2_gw = [&](int i) { return (const float*)d_in[MB + 4 + 3 * i + 1]; };
  auto m2m_ctr2_w  = [&](int i) { return (const float*)d_in[MB + 4 + 3 * i + 2]; };
  auto m2m_left    = [&](int i) { return (const float*)d_in[MB + 16 + i]; };
  auto m2m_ngb     = [&](int i) { return (const float*)d_in[MB + 20 + i]; };
  auto m2m_ngw     = [&](int i) { return (const float*)d_in[MB + 24 + i]; };
  auto m2m_pre     = [&](int i, int sc) { return (const float*)d_in[MB + 28 + 6 * i + sc]; };
  auto m2m_right   = [&](int i) { return (const float*)d_in[MB + 52 + i]; };
  auto m2m_suc     = [&](int i, int sc) { return (const float*)d_in[MB + 56 + 6 * i + sc]; };

  Ws w = carve(d_ws);

  // ================= A2M =================
  build_edges(s, w, graph_ctrs, N_MAP, actor_ctrs, N_ACTS, 7.0f, E_A2M);
  concat_meta_kernel<<<(N_MAP * 132 + 255) / 256, 256, 0, s>>>(
      nodes, graph_turn, graph_ctrl, graph_inter, w.nm);
  { GArgs a; a.X0 = w.nm; a.k0 = 132; a.W = meta_w; a.gw = meta_gw; a.gb = meta_gb;
    a.out = w.nb0; a.M = N_MAP; launch_gemm(s, G_GN_ACT, a); }
  run_att(s, w, a2m0, w.nb0, N_MAP, actors, E_A2M, w.nb1, w.nb2, w.nb1);
  run_att(s, w, a2m1, w.nb1, N_MAP, actors, E_A2M, w.nb2, w.nb0, w.nb2);

  // ================= M2M =================
  float* cur = w.nb2;   // feat
  float* fA  = w.nb0;   // temp / next-feat
  float* fB  = w.nb1;   // gn scratch
  for (int i = 0; i < 4; ++i) {
    { GArgs a; a.X0 = cur; a.k0 = NF; a.W = m2m_ctr(i); a.out = fA; a.M = N_MAP;
      launch_gemm(s, G_PLAIN, a); }
    for (int sc = 0; sc < 6; ++sc) {
      { GArgs a; a.X0 = cur; a.I0 = pre_v + sc * N_MAP; a.k0 = NF;
        a.W = m2m_pre(i, sc); a.oidx = pre_u + sc * N_MAP; a.out = fA; a.M = N_MAP;
        launch_gemm(s, G_SCATTER, a); }
      { GArgs a; a.X0 = cur; a.I0 = suc_v + sc * N_MAP; a.k0 = NF;
        a.W = m2m_suc(i, sc); a.oidx = suc_u + sc * N_MAP; a.out = fA; a.M = N_MAP;
        launch_gemm(s, G_SCATTER, a); }
    }
    { GArgs a; a.X0 = cur; a.I0 = left_v; a.k0 = NF; a.W = m2m_left(i);
      a.oidx = left_u; a.out = fA; a.M = E_LR; launch_gemm(s, G_SCATTER, a); }
    { GArgs a; a.X0 = cur; a.I0 = right_v; a.k0 = NF; a.W = m2m_right(i);
      a.oidx = right_u; a.out = fA; a.M = E_LR; launch_gemm(s, G_SCATTER, a); }
    gn_relu_kernel<<<N_MAP, 128, 0, s>>>(fA, m2m_ngw(i), m2m_ngb(i), fB);
    { GArgs a; a.X0 = fB; a.k0 = NF; a.W = m2m_ctr2_w(i); a.gw = m2m_ctr2_gw(i);
      a.gb = m2m_ctr2_gb(i); a.res = cur; a.out = fA; a.M = N_MAP;
      launch_gemm(s, G_GN_RES_ACT, a); }
    float* t = cur; cur = fA; fA = t;   // rotate
  }
  const float* feat = cur;              // == w.nb2 after 4 fuses

  // ================= M2A =================
  build_edges(s, w, actor_ctrs, N_ACTS, graph_ctrs, N_MAP, 6.0f, E_M2A);
  run_att(s, w, m2a0, actors, N_ACTS, feat, E_M2A, w.ab0, w.ab1, w.ab0);
  run_att(s, w, m2a1, w.ab0,  N_ACTS, feat, E_M2A, w.ab1, w.ab2, w.ab1);

  // ================= A2A =================
  build_edges(s, w, actor_ctrs, N_ACTS, actor_ctrs, N_ACTS, 100.0f, E_A2A);
  run_att(s, w, a2a0, w.ab1, N_ACTS, w.ab1, E_A2A, w.ab2, w.ab0, w.ab2);
  run_att(s, w, a2a1, w.ab2, N_ACTS, w.ab2, E_A2A, w.ab0, w.ab1, (float*)d_out);
}
